// MultiHead_Local_Global_Attention_22033182228931
// MI455X (gfx1250) — compile-verified, matrix operand order checked by reading, not run on hardware
//
#include <hip/hip_runtime.h>
#include <hip/hip_bf16.h>

// ---------------- problem constants ----------------
#define BATCH   2
#define SEQ     2048
#define DM      1024      // D_IN == D_OUT
#define NH      16
#define DHEAD   64
#define WIN     64
#define MAXG    256       // max tracked global tokens per batch (reference has 8)
#define BN      (BATCH*SEQ)

typedef __bf16 bf16_t;
typedef __attribute__((ext_vector_type(16))) __bf16 v16bf;
typedef __attribute__((ext_vector_type(8)))  float  v8f;

struct alignas(16) Pack16 { unsigned int w[4]; };   // 8 bf16 = 16 bytes

static __device__ inline v8f v8zero() {
  v8f z;
  #pragma unroll
  for (int i = 0; i < 8; ++i) z[i] = 0.f;
  return z;
}

static __device__ inline v8f wmma_bf16(v16bf a, v16bf b, v8f c) {
  return __builtin_amdgcn_wmma_f32_16x16x32_bf16(false, a, false, b, (short)0, c,
                                                 false, false);
}

// Load a 16x32 bf16 fragment (A layout; B layout is the symmetric transpose:
// lane-index selects the 16-row/col, per-lane contiguous run covers 8 of the
// 32 K values twice).  Per ISA 7.12.2: lanes 0-15 hold K 0..7 / 16..23,
// lanes 16-31 hold K 8..15 / 24..31, 2 values per VGPR.
static __device__ inline v16bf load_frag(const bf16_t* base, int ld,
                                         int row_base, int k_base, int lane) {
  const int l    = lane & 15;
  const int half = lane >> 4;
  const bf16_t* p = base + (size_t)(row_base + l) * ld + k_base + half * 8;
  union { v16bf v; Pack16 h[2]; } u;
  u.h[0] = *(const Pack16*)(p);
  u.h[1] = *(const Pack16*)(p + 16);
  return u.v;
}

// =====================================================================
// GEMM: C[M,Nd] = A[M,Kd] * Bw[Kd,Nd]  (A f32 or bf16, Bw f32, bf16 WMMA)
// outMode: 0 = bf16 row-major, 1 = bf16 transposed per batch ([B,DM,SEQ]),
//          2 = f32 row-major
// =====================================================================
#define TM 128
#define TN 128
#define TK 32

__global__ __launch_bounds__(256)
void gemm_bf16_kernel(const void* __restrict__ Aptr, int aIsBf16,
                      const float* __restrict__ Bw,
                      void* __restrict__ Out, int outMode,
                      int M, int Kd, int Nd) {
  __shared__ alignas(16) bf16_t As[TM][TK];
  __shared__ alignas(16) bf16_t Bs[TN][TK];   // stored transposed: Bs[col][k]

  const int tid  = threadIdx.x;
  const int lane = tid & 31;
  const int wv   = tid >> 5;
  const int wm   = wv & 3;     // 4 row groups of 32
  const int wn   = wv >> 2;    // 2 col groups of 64
  const int rowT = blockIdx.y * TM;
  const int colT = blockIdx.x * TN;

  const float*  Af = (const float*)Aptr;
  const bf16_t* Ab = (const bf16_t*)Aptr;

  v8f acc[2][4];
  #pragma unroll
  for (int i = 0; i < 2; ++i)
    #pragma unroll
    for (int j = 0; j < 4; ++j) acc[i][j] = v8zero();

  for (int kc = 0; kc < Kd; kc += TK) {
    // ---- stage A tile (convert to bf16 if needed) ----
    if (aIsBf16) {
      #pragma unroll
      for (int it = 0; it < 2; ++it) {
        int c  = tid + it * 256;          // 512 chunks of 8 bf16
        int r  = c >> 2;
        int k8 = (c & 3) * 8;
        *(Pack16*)&As[r][k8] =
            *(const Pack16*)(Ab + (size_t)(rowT + r) * Kd + kc + k8);
      }
    } else {
      #pragma unroll
      for (int it = 0; it < 4; ++it) {
        int u  = tid + it * 256;          // 1024 float4 units
        int r  = u >> 3;
        int k4 = (u & 7) * 4;
        const float4 x = *(const float4*)(Af + (size_t)(rowT + r) * Kd + kc + k4);
        As[r][k4 + 0] = (bf16_t)x.x;  As[r][k4 + 1] = (bf16_t)x.y;
        As[r][k4 + 2] = (bf16_t)x.z;  As[r][k4 + 3] = (bf16_t)x.w;
      }
    }
    // ---- stage B tile transposed: Bs[c][k] = Bw[kc+k][colT+c] ----
    #pragma unroll
    for (int it = 0; it < 16; ++it) {
      int u = tid + it * 256;             // 4096 elements
      int k = u >> 7;
      int c = u & 127;
      Bs[c][k] = (bf16_t)Bw[(size_t)(kc + k) * Nd + colT + c];
    }
    if (kc + TK < Kd)   // hint next weight slab into cache (global_prefetch_b8)
      __builtin_prefetch(Bw + (size_t)(kc + TK) * Nd + colT + (tid & 127), 0, 1);
    __syncthreads();

    v16bf af[2], bfg[4];
    #pragma unroll
    for (int fm = 0; fm < 2; ++fm)
      af[fm] = load_frag(&As[0][0], TK, wm * 32 + fm * 16, 0, lane);
    #pragma unroll
    for (int fn = 0; fn < 4; ++fn)
      bfg[fn] = load_frag(&Bs[0][0], TK, wn * 64 + fn * 16, 0, lane);
    #pragma unroll
    for (int fm = 0; fm < 2; ++fm)
      #pragma unroll
      for (int fn = 0; fn < 4; ++fn)
        acc[fm][fn] = wmma_bf16(af[fm], bfg[fn], acc[fm][fn]);
    __syncthreads();
  }

  // ---- store ----
  const int nl = lane & 15, half = lane >> 4;
  #pragma unroll
  for (int fm = 0; fm < 2; ++fm)
    #pragma unroll
    for (int fn = 0; fn < 4; ++fn)
      #pragma unroll
      for (int r = 0; r < 8; ++r) {
        int rowG = rowT + wm * 32 + fm * 16 + r + 8 * half;
        int colG = colT + wn * 64 + fn * 16 + nl;
        float v  = acc[fm][fn][r];
        if (outMode == 0) {
          ((bf16_t*)Out)[(size_t)rowG * Nd + colG] = (bf16_t)v;
        } else if (outMode == 1) {           // per-batch transpose: [B, DM, SEQ]
          int bb = rowG >> 11;               // SEQ == 2048
          int nn = rowG & (SEQ - 1);
          ((bf16_t*)Out)[((size_t)bb * Nd + colG) * SEQ + nn] = (bf16_t)v;
        } else {
          ((float*)Out)[(size_t)rowG * Nd + colG] = v;
        }
      }
}

// =====================================================================
// Deterministic global-token discovery (positions via prefix counts)
// =====================================================================
__global__ void find_globals_kernel(const unsigned char* __restrict__ g,
                                    int* __restrict__ gidx,
                                    int* __restrict__ gcnt) {
  int b = blockIdx.x;
  const unsigned char* gb = g + (size_t)b * SEQ;
  if (threadIdx.x == 0) {
    int c = 0;
    for (int i = 0; i < SEQ; ++i) c += gb[i] ? 1 : 0;
    gcnt[b] = c < MAXG ? c : MAXG;
  }
  for (int i = threadIdx.x; i < SEQ; i += blockDim.x) {
    if (gb[i]) {
      int p = 0;
      for (int j = 0; j < i; ++j) p += gb[j] ? 1 : 0;
      if (p < MAXG) gidx[b * MAXG + p] = i;
    }
  }
}

// Gather global K rows (row-major) and global V columns (transposed, ld=MAXG)
__global__ void gather_globals_kernel(const bf16_t* __restrict__ Kmat,
                                      const bf16_t* __restrict__ Vt,
                                      const int* __restrict__ gidx,
                                      const int* __restrict__ gcnt,
                                      bf16_t* __restrict__ Kg,
                                      bf16_t* __restrict__ VgT) {
  int slot = blockIdx.x, b = blockIdx.y;
  if (slot >= gcnt[b]) return;
  int idx = gidx[b * MAXG + slot];
  const bf16_t* src = Kmat + ((size_t)b * SEQ + idx) * DM;
  bf16_t* dst = Kg + ((size_t)b * MAXG + slot) * DM;
  for (int d = threadIdx.x; d < DM; d += blockDim.x) dst[d] = src[d];
  for (int d = threadIdx.x; d < DM; d += blockDim.x)
    VgT[((size_t)b * DM + d) * MAXG + slot] = Vt[((size_t)b * DM + d) * SEQ + idx];
}

// =====================================================================
// Flash-style local+global-column attention (one wave per 16-query tile)
// =====================================================================
static __device__ inline void softmax_tile_update(const v8f& s0, const v8f& s1,
                                                  const bool* okA, const bool* okB,
                                                  float* m, float* l, v8f* o,
                                                  bf16_t* Pw, int lane) {
  const int n = lane & 15, half = lane >> 4;
  #pragma unroll
  for (int r = 0; r < 8; ++r) {
    float a  = okA[r] ? s0[r] * 0.125f : -1e30f;
    float bb = okB[r] ? s1[r] * 0.125f : -1e30f;
    float t = fmaxf(a, bb);
    t = fmaxf(t, __shfl_xor(t, 1, 32));
    t = fmaxf(t, __shfl_xor(t, 2, 32));
    t = fmaxf(t, __shfl_xor(t, 4, 32));
    t = fmaxf(t, __shfl_xor(t, 8, 32));
    float mn = fmaxf(m[r], t);
    float al = __expf(m[r] - mn);
    float p0 = okA[r] ? __expf(a - mn)  : 0.f;
    float p1 = okB[r] ? __expf(bb - mn) : 0.f;
    float rs = p0 + p1;
    rs += __shfl_xor(rs, 1, 32);
    rs += __shfl_xor(rs, 2, 32);
    rs += __shfl_xor(rs, 4, 32);
    rs += __shfl_xor(rs, 8, 32);
    l[r] = l[r] * al + rs;
    #pragma unroll
    for (int f = 0; f < 4; ++f) o[f][r] *= al;
    m[r] = mn;
    Pw[(r + 8 * half) * 32 + n]      = (bf16_t)p0;
    Pw[(r + 8 * half) * 32 + 16 + n] = (bf16_t)p1;
  }
}

__global__ __launch_bounds__(128)
void attn_local_kernel(const bf16_t* __restrict__ Q,
                       const bf16_t* __restrict__ Kmat,
                       const bf16_t* __restrict__ Vt,
                       const bf16_t* __restrict__ Kg,
                       const bf16_t* __restrict__ VgT,
                       const unsigned char* __restrict__ g,
                       const int* __restrict__ gcnt,
                       bf16_t* __restrict__ ctx) {
  __shared__ alignas(16) bf16_t Pbuf[4][16 * 32];

  const int lane = threadIdx.x & 31;
  const int wv   = threadIdx.x >> 5;
  const int h = blockIdx.y, b = blockIdx.z;
  const int blockQ = blockIdx.x * 64;
  const int qBase  = blockQ + wv * 16;
  const int n = lane & 15, half = lane >> 4;
  const int hb = h * DHEAD;

  const bf16_t* Qb  = Q    + (size_t)b * SEQ * DM;
  const bf16_t* Kb  = Kmat + (size_t)b * SEQ * DM;
  const bf16_t* Vtb = Vt   + (size_t)b * DM * SEQ;
  const bf16_t* Kgb = Kg   + (size_t)b * MAXG * DM;
  const bf16_t* Vgb = VgT  + (size_t)b * DM * MAXG;
  const unsigned char* gb = g + (size_t)b * SEQ;
  bf16_t* Pw = &Pbuf[wv][0];

  v16bf qf0 = load_frag(Qb, DM, qBase, hb,      lane);
  v16bf qf1 = load_frag(Qb, DM, qBase, hb + 32, lane);

  float m[8], l[8];
  v8f o[4];
  #pragma unroll
  for (int r = 0; r < 8; ++r) { m[r] = -1e30f; l[r] = 0.f; }
  #pragma unroll
  for (int f = 0; f < 4; ++f) o[f] = v8zero();

  // ---------- causal window pass (global key-columns excluded) ----------
  const int start = blockQ >= 64 ? blockQ - 64 : 0;
  for (int j0 = start; j0 <= blockQ + 63; j0 += 32) {
    v8f s0 = v8zero(), s1 = v8zero();
    s0 = wmma_bf16(qf0, load_frag(Kb, DM, j0,      hb,      lane), s0);
    s0 = wmma_bf16(qf1, load_frag(Kb, DM, j0,      hb + 32, lane), s0);
    s1 = wmma_bf16(qf0, load_frag(Kb, DM, j0 + 16, hb,      lane), s1);
    s1 = wmma_bf16(qf1, load_frag(Kb, DM, j0 + 16, hb + 32, lane), s1);

    const int jA = j0 + n, jB = j0 + 16 + n;
    const bool gA = gb[jA] != 0, gB = gb[jB] != 0;
    bool okA[8], okB[8];
    #pragma unroll
    for (int r = 0; r < 8; ++r) {
      int i = qBase + r + 8 * half;
      okA[r] = (jA <= i) && (i - jA < WIN) && !gA;
      okB[r] = (jB <= i) && (i - jB < WIN) && !gB;
    }
    softmax_tile_update(s0, s1, okA, okB, m, l, o, Pw, lane);
    __syncthreads();
    {
      v16bf pf = load_frag(Pw, 32, 0, 0, lane);
      #pragma unroll
      for (int f = 0; f < 4; ++f)
        o[f] = wmma_bf16(pf, load_frag(Vtb, SEQ, hb + f * 16, j0, lane), o[f]);
    }
    __syncthreads();
  }

  // ---------- gathered global key columns (unmasked for every row) ----------
  const int gc = gcnt[b];
  for (int g0 = 0; g0 < gc; g0 += 32) {
    v8f s0 = v8zero(), s1 = v8zero();
    s0 = wmma_bf16(qf0, load_frag(Kgb, DM, g0,      hb,      lane), s0);
    s0 = wmma_bf16(qf1, load_frag(Kgb, DM, g0,      hb + 32, lane), s0);
    s1 = wmma_bf16(qf0, load_frag(Kgb, DM, g0 + 16, hb,      lane), s1);
    s1 = wmma_bf16(qf1, load_frag(Kgb, DM, g0 + 16, hb + 32, lane), s1);
    bool okA[8], okB[8];
    const bool vA = (g0 + n) < gc, vB = (g0 + 16 + n) < gc;
    #pragma unroll
    for (int r = 0; r < 8; ++r) { okA[r] = vA; okB[r] = vB; }
    softmax_tile_update(s0, s1, okA, okB, m, l, o, Pw, lane);
    __syncthreads();
    {
      v16bf pf = load_frag(Pw, 32, 0, 0, lane);
      #pragma unroll
      for (int f = 0; f < 4; ++f)
        o[f] = wmma_bf16(pf, load_frag(Vgb, MAXG, hb + f * 16, g0, lane), o[f]);
    }
    __syncthreads();
  }

  // ---------- normalize + store ctx ([B, N, H*DH] row-major, bf16) ----------
  #pragma unroll
  for (int r = 0; r < 8; ++r) {
    float inv = 1.f / l[r];
    int i = qBase + r + 8 * half;
    #pragma unroll
    for (int f = 0; f < 4; ++f)
      ctx[((size_t)b * SEQ + i) * DM + hb + f * 16 + n] = (bf16_t)(o[f][r] * inv);
  }
}

// =====================================================================
// Global query rows: full (unmasked) attention over all keys; overwrites
// the rows the local kernel computed.  Deterministic serial accumulation.
// =====================================================================
__global__ __launch_bounds__(256)
void attn_global_rows_kernel(const bf16_t* __restrict__ Q,
                             const bf16_t* __restrict__ Kmat,
                             const bf16_t* __restrict__ Vt,
                             const int* __restrict__ gidx,
                             const int* __restrict__ gcnt,
                             bf16_t* __restrict__ ctx) {
  const int slot = blockIdx.x, h = blockIdx.y, b = blockIdx.z;
  if (slot >= gcnt[b]) return;
  const int i  = gidx[b * MAXG + slot];
  const int hb = h * DHEAD;
  const int tid = threadIdx.x;

  __shared__ float qs[DHEAD];
  __shared__ float sc[SEQ];
  __shared__ float red[256];

  const bf16_t* Qrow = Q + ((size_t)b * SEQ + i) * DM + hb;
  if (tid < DHEAD) qs[tid] = (float)Qrow[tid];
  __syncthreads();

  float lmax = -1e30f;
  for (int j = tid; j < SEQ; j += 256) {
    const bf16_t* krow = Kmat + ((size_t)b * SEQ + j) * DM + hb;
    float s = 0.f;
    #pragma unroll 8
    for (int d = 0; d < DHEAD; ++d) s += qs[d] * (float)krow[d];
    s *= 0.125f;
    sc[j] = s;
    lmax = fmaxf(lmax, s);
  }
  red[tid] = lmax; __syncthreads();
  for (int s2 = 128; s2 > 0; s2 >>= 1) {
    if (tid < s2) red[tid] = fmaxf(red[tid], red[tid + s2]);
    __syncthreads();
  }
  const float gmax = red[0];
  __syncthreads();

  float lsum = 0.f;
  for (int j = tid; j < SEQ; j += 256) {
    float p = __expf(sc[j] - gmax);
    sc[j] = p;
    lsum += p;
  }
  red[tid] = lsum; __syncthreads();
  for (int s2 = 128; s2 > 0; s2 >>= 1) {
    if (tid < s2) red[tid] += red[tid + s2];
    __syncthreads();
  }
  const float inv = 1.f / red[0];
  __syncthreads();

  if (tid < DHEAD) {
    const bf16_t* vrow = Vt + ((size_t)b * DM + hb + tid) * SEQ;
    float acc = 0.f;
    for (int j = 0; j < SEQ; ++j) acc += sc[j] * (float)vrow[j];
    ctx[((size_t)b * SEQ + i) * DM + hb + tid] = (bf16_t)(acc * inv);
  }
}

// =====================================================================
// launch
// =====================================================================
extern "C" void kernel_launch(void* const* d_in, const int* in_sizes, int n_in,
                              void* d_out, int out_size, void* d_ws, size_t ws_size,
                              hipStream_t stream) {
  const float* x  = (const float*)d_in[0];
  const float* wq = (const float*)d_in[1];
  const float* wk = (const float*)d_in[2];
  const float* wv = (const float*)d_in[3];
  const float* wo = (const float*)d_in[4];
  const unsigned char* gatt = (const unsigned char*)d_in[5];
  float* out = (float*)d_out;

  char* w = (char*)d_ws;
  size_t off = 0;
  const size_t szQKV = (size_t)BN * DM * sizeof(bf16_t);   // 8 MB each
  bf16_t* Qm  = (bf16_t*)(w + off); off += szQKV;
  bf16_t* Km  = (bf16_t*)(w + off); off += szQKV;
  bf16_t* Vt  = (bf16_t*)(w + off); off += szQKV;
  bf16_t* Ctx = (bf16_t*)(w + off); off += szQKV;
  bf16_t* Kg  = (bf16_t*)(w + off); off += (size_t)BATCH * MAXG * DM * sizeof(bf16_t);
  bf16_t* VgT = (bf16_t*)(w + off); off += (size_t)BATCH * DM * MAXG * sizeof(bf16_t);
  int* gidx   = (int*)(w + off);    off += (size_t)BATCH * MAXG * sizeof(int);
  int* gcnt   = (int*)(w + off);    off += (size_t)BATCH * sizeof(int);
  (void)ws_size; (void)in_sizes; (void)n_in; (void)out_size;

  dim3 ggrid(DM / TN, BN / TM);

  // Q, K row-major bf16; V transposed bf16 [B, DM, SEQ]
  gemm_bf16_kernel<<<ggrid, 256, 0, stream>>>(x, 0, wq, Qm, 0, BN, DM, DM);
  gemm_bf16_kernel<<<ggrid, 256, 0, stream>>>(x, 0, wk, Km, 0, BN, DM, DM);
  gemm_bf16_kernel<<<ggrid, 256, 0, stream>>>(x, 0, wv, Vt, 1, BN, DM, DM);

  find_globals_kernel<<<BATCH, 256, 0, stream>>>(gatt, gidx, gcnt);
  gather_globals_kernel<<<dim3(MAXG, BATCH), 256, 0, stream>>>(Km, Vt, gidx, gcnt,
                                                               Kg, VgT);

  attn_local_kernel<<<dim3(SEQ / 64, NH, BATCH), 128, 0, stream>>>(
      Qm, Km, Vt, Kg, VgT, gatt, gcnt, Ctx);
  attn_global_rows_kernel<<<dim3(MAXG, NH, BATCH), 256, 0, stream>>>(
      Qm, Km, Vt, gidx, gcnt, Ctx);

  // out = Ctx @ wo (f32 output)
  gemm_bf16_kernel<<<ggrid, 256, 0, stream>>>(Ctx, 1, wo, out, 2, BN, DM, DM);
}
